// VectorQuantizer_33526514712760
// MI455X (gfx1250) — compile-verified, matrix operand order checked by reading, not run on hardware
//
#include <hip/hip_runtime.h>
#include <hip/hip_bf16.h>
#include <float.h>

// MI455X / gfx1250 VQ kernel: split-precision bf16 WMMA GEMM + fused argmin,
// double-buffered LDS staging via GLOBAL_LOAD_ASYNC_TO_LDS (ASYNCcnt path),
// dual accumulator chains to break WMMA->WMMA RAW hazards.

typedef __attribute__((ext_vector_type(16))) __bf16 v16bf;
typedef __attribute__((ext_vector_type(8)))  float  v8f;
typedef __attribute__((ext_vector_type(4)))  int    v4i;

#define BDIM   16
#define CDIM   256
#define TDIM   2048
#define KCODES 8192
#define NROWS  (BDIM * TDIM)          // 32768
#define CT     (CDIM * TDIM)          // 524288
#define BCT    (BDIM * CDIM * TDIM)   // 8388608

#if defined(__gfx1250__) && __has_builtin(__builtin_amdgcn_global_load_async_to_lds_b128) && \
    __has_builtin(__builtin_amdgcn_s_wait_asynccnt)
#define VQ_ASYNC_COPY 1
typedef __attribute__((address_space(1))) v4i* vq_gptr;   // global (AS1)
typedef __attribute__((address_space(3))) v4i* vq_lptr;   // LDS (AS3)
#else
#define VQ_ASYNC_COPY 0
#endif

__device__ __forceinline__ unsigned short f32_to_bf16_rn(float f) {
  unsigned u = __float_as_uint(f);
  u += 0x7FFFu + ((u >> 16) & 1u);
  return (unsigned short)(u >> 16);
}
__device__ __forceinline__ float bf16_bits_to_f32(unsigned short h) {
  return __uint_as_float(((unsigned)h) << 16);
}

// Stage one 32-row codebook chunk (hi+lo, 16 KB each) into an LDS buffer,
// padded row pitch 264 shorts (33 uint4) for bank-conflict-free b128 reads.
__device__ __forceinline__ void vq_stage_chunk(const unsigned short* __restrict__ src_hi,
                                               const unsigned short* __restrict__ src_lo,
                                               uint4* dst_hi, uint4* dst_lo, int tid) {
#pragma unroll
  for (int i = 0; i < 4; ++i) {
    const int e = tid + i * 256;             // uint4 index 0..1023 of the chunk
    const int d = (e >> 5) * 33 + (e & 31);  // padded LDS layout
    const uint4* gh = ((const uint4*)src_hi) + e;
    const uint4* gl = ((const uint4*)src_lo) + e;
#if VQ_ASYNC_COPY
    // Direct memory->LDS DMA, no VGPR round-trip; tracked by ASYNCcnt.
    __builtin_amdgcn_global_load_async_to_lds_b128(
        (vq_gptr)(unsigned long long)(uintptr_t)gh,
        (vq_lptr)(unsigned)(uintptr_t)(dst_hi + d), 0, 0);
    __builtin_amdgcn_global_load_async_to_lds_b128(
        (vq_gptr)(unsigned long long)(uintptr_t)gl,
        (vq_lptr)(unsigned)(uintptr_t)(dst_lo + d), 0, 0);
#else
    dst_hi[d] = *gh;
    dst_lo[d] = *gl;
#endif
  }
}

// ---------------------------------------------------------------------------
// Kernel 0: split codebook into bf16 hi/lo pair and compute row norms ||e||^2.
// One wave per codebook row; lane handles 8 contiguous floats.
// ---------------------------------------------------------------------------
__global__ void __launch_bounds__(256)
vq_prep_codebook(const float* __restrict__ cb,
                 unsigned short* __restrict__ cb_hi,
                 unsigned short* __restrict__ cb_lo,
                 float* __restrict__ norms) {
  const int wave = threadIdx.x >> 5;
  const int lane = threadIdx.x & 31;
  const int k = blockIdx.x * 8 + wave;
  const float* row = cb + (size_t)k * CDIM;
  const int c0 = lane * 8;

  union { unsigned short s[8]; uint4 q; } uh, ul;
  float s = 0.f;
#pragma unroll
  for (int j = 0; j < 8; ++j) {
    float f = row[c0 + j];
    s += f * f;
    unsigned short h = f32_to_bf16_rn(f);
    uh.s[j] = h;
    ul.s[j] = f32_to_bf16_rn(f - bf16_bits_to_f32(h));
  }
  *(uint4*)(cb_hi + (size_t)k * CDIM + c0) = uh.q;
  *(uint4*)(cb_lo + (size_t)k * CDIM + c0) = ul.q;

#pragma unroll
  for (int m = 16; m >= 1; m >>= 1) s += __shfl_xor(s, m, 32);
  if (lane == 0) norms[k] = s;
}

// ---------------------------------------------------------------------------
// Kernel 1: fused GEMM (z @ codebook^T via bf16x3 WMMA) + row argmin.
// Block = 8 waves; each wave owns 16 rows (persistent A frags); all waves
// share double-buffered 32-code LDS chunks. Rank by ||e||^2 - 2*z.e.
// ---------------------------------------------------------------------------
__global__ void __launch_bounds__(256)
vq_argmin(const float* __restrict__ z,
          const unsigned short* __restrict__ cb_hi,
          const unsigned short* __restrict__ cb_lo,
          const float* __restrict__ norms,
          int* __restrict__ out_idx) {
  __shared__ uint4 sh_hi[2][32 * 33];   // 2 x 16.5 KB
  __shared__ uint4 sh_lo[2][32 * 33];

  const int tid = threadIdx.x;
  const int lane = tid & 31;
  const int wave = tid >> 5;
  const int m0 = blockIdx.x * 128 + wave * 16;
  const int halfsel = lane >> 4;     // 0: K elems {0..7,16..23}; 1: {8..15,24..31}
  const int l15 = lane & 15;

  // ---- Build persistent A fragments (16 z rows, C=256 -> 8 K-chunks) ----
  const int mrow = m0 + l15;               // row n = b*T + t
  const int bb = mrow >> 11;               // / TDIM
  const int tt = mrow & (TDIM - 1);
  const float* zb = z + (size_t)bb * CT + tt;   // element c is zb[c * TDIM]

  v16bf ahi[8], alo[8];
#pragma unroll
  for (int kc = 0; kc < 8; ++kc) {
    const int cbase = kc * 32 + halfsel * 8;
    union { unsigned short s[16]; v16bf v; } uh, ul;
#pragma unroll
    for (int e = 0; e < 16; ++e) {
      const int c = cbase + (e < 8 ? e : e + 8);   // {0..7} then {16..23}
      float f = zb[(size_t)c * TDIM];
      unsigned short h = f32_to_bf16_rn(f);
      uh.s[e] = h;
      ul.s[e] = f32_to_bf16_rn(f - bf16_bits_to_f32(h));
    }
    ahi[kc] = uh.v;
    alo[kc] = ul.v;
  }

  float minval[8];
  int   minidx[8];
#pragma unroll
  for (int r = 0; r < 8; ++r) { minval[r] = FLT_MAX; minidx[r] = 0; }

  // Prologue: stage chunk 0 into buffer 0.
  vq_stage_chunk(cb_hi, cb_lo, sh_hi[0], sh_lo[0], tid);

  for (int k0 = 0; k0 < KCODES; k0 += 32) {
    const int cur = (k0 >> 5) & 1;
#if VQ_ASYNC_COPY
    __builtin_amdgcn_s_wait_asynccnt(0);   // own slice of chunk k0 landed
#endif
    __syncthreads();                       // all waves' slices landed

    // Prefetch next chunk into the other buffer (overlaps with WMMA below).
    if (k0 + 32 < KCODES) {
      vq_stage_chunk(cb_hi + (size_t)(k0 + 32) * CDIM,
                     cb_lo + (size_t)(k0 + 32) * CDIM,
                     sh_hi[cur ^ 1], sh_lo[cur ^ 1], tid);
    }

    unsigned short* shp_hi = (unsigned short*)sh_hi[cur];
    unsigned short* shp_lo = (unsigned short*)sh_lo[cur];

    // Two independent accumulator chains (code columns k0+0..15, k0+16..31):
    // alternating WMMAs breaks the WMMA->WMMA RAW hazard chain.
    const int ncol0 = k0 + l15;
    const int ncol1 = k0 + 16 + l15;
    const float nrm0 = norms[ncol0];
    const float nrm1 = norms[ncol1];
    v8f acc0 = {};
    v8f acc1 = {};
#pragma unroll
    for (int kc = 0; kc < 8; ++kc) {
      const int coff = kc * 32 + halfsel * 8;
      const int off0 = l15 * 264 + coff;          // code rows 0..15 of chunk
      const int off1 = (16 + l15) * 264 + coff;   // code rows 16..31 of chunk
      union { uint4 q[2]; v16bf v; } bh0, bl0, bh1, bl1;
      bh0.q[0] = *(const uint4*)(shp_hi + off0);
      bh0.q[1] = *(const uint4*)(shp_hi + off0 + 16);
      bh1.q[0] = *(const uint4*)(shp_hi + off1);
      bh1.q[1] = *(const uint4*)(shp_hi + off1 + 16);
      bl0.q[0] = *(const uint4*)(shp_lo + off0);
      bl0.q[1] = *(const uint4*)(shp_lo + off0 + 16);
      bl1.q[0] = *(const uint4*)(shp_lo + off1);
      bl1.q[1] = *(const uint4*)(shp_lo + off1 + 16);
      // dot ~= Ahi*Bhi + Ahi*Blo + Alo*Bhi  (fp32-grade accuracy)
      acc0 = __builtin_amdgcn_wmma_f32_16x16x32_bf16(
          false, ahi[kc], false, bh0.v, (short)0, acc0, false, false);
      acc1 = __builtin_amdgcn_wmma_f32_16x16x32_bf16(
          false, ahi[kc], false, bh1.v, (short)0, acc1, false, false);
      acc0 = __builtin_amdgcn_wmma_f32_16x16x32_bf16(
          false, ahi[kc], false, bl0.v, (short)0, acc0, false, false);
      acc1 = __builtin_amdgcn_wmma_f32_16x16x32_bf16(
          false, ahi[kc], false, bl1.v, (short)0, acc1, false, false);
      acc0 = __builtin_amdgcn_wmma_f32_16x16x32_bf16(
          false, alo[kc], false, bh0.v, (short)0, acc0, false, false);
      acc1 = __builtin_amdgcn_wmma_f32_16x16x32_bf16(
          false, alo[kc], false, bh1.v, (short)0, acc1, false, false);
    }
#pragma unroll
    for (int r = 0; r < 8; ++r) {
      const float d0 = nrm0 - 2.0f * acc0[r];
      if (d0 < minval[r]) { minval[r] = d0; minidx[r] = ncol0; }
      const float d1 = nrm1 - 2.0f * acc1[r];
      if (d1 < minval[r]) { minval[r] = d1; minidx[r] = ncol1; }
    }
  }

  // Cross-lane argmin within each 16-lane half (lanes 0-15: rows m0+r,
  // lanes 16-31: rows m0+8+r). Ties resolve to the smaller index (jnp argmin).
#pragma unroll
  for (int r = 0; r < 8; ++r) {
    float v = minval[r];
    int ix = minidx[r];
#pragma unroll
    for (int m = 8; m >= 1; m >>= 1) {
      float ov = __shfl_xor(v, m, 32);
      int   oi = __shfl_xor(ix, m, 32);
      if (ov < v || (ov == v && oi < ix)) { v = ov; ix = oi; }
    }
    minidx[r] = ix;
  }
  if (lane == 0) {
#pragma unroll
    for (int r = 0; r < 8; ++r) out_idx[m0 + r] = minidx[r];
  } else if (lane == 16) {
#pragma unroll
    for (int r = 0; r < 8; ++r) out_idx[m0 + 8 + r] = minidx[r];
  }
}

// ---------------------------------------------------------------------------
// Kernel 2: gather z_q = codebook[idx], write z_q_ste (== z_q numerically),
// float-cast indices, and per-block loss partial sums.
// Block covers (b, 8 channels, 256 consecutive t) -> coalesced z/out access.
// ---------------------------------------------------------------------------
__global__ void __launch_bounds__(256)
vq_gather(const float* __restrict__ z,
          const float* __restrict__ cb,
          const int* __restrict__ idx,
          float* __restrict__ out_zq,
          float* __restrict__ out_idx_f,
          float* __restrict__ partials) {
  __shared__ float red[8];
  const int tid = threadIdx.x;
  const int b   = blockIdx.x >> 8;
  const int rem = blockIdx.x & 255;
  const int c0  = (rem >> 3) * 8;
  const int t   = (rem & 7) * 256 + tid;
  const int n   = b * TDIM + t;
  const int ix  = idx[n];
  const float* cbr = cb + (size_t)ix * CDIM;

  float s = 0.f;
#pragma unroll
  for (int j = 0; j < 8; ++j) {
    const int c = c0 + j;
    const size_t o = (size_t)b * CT + (size_t)c * TDIM + t;
    const float q  = cbr[c];
    const float zv = z[o];
    out_zq[o] = q;                       // z + sg(z_q - z) == z_q
    const float d = q - zv;
    s += d * d;
  }
  if (c0 == 0) out_idx_f[n] = (float)ix;

#pragma unroll
  for (int m = 16; m >= 1; m >>= 1) s += __shfl_xor(s, m, 32);
  if ((tid & 31) == 0) red[tid >> 5] = s;
  __syncthreads();
  if (tid == 0) {
    float tot = 0.f;
#pragma unroll
    for (int w = 0; w < 8; ++w) tot += red[w];
    partials[blockIdx.x] = tot;
  }
}

// ---------------------------------------------------------------------------
// Kernel 3: reduce loss partials -> mean (overwrites output; replay-safe).
// ---------------------------------------------------------------------------
__global__ void __launch_bounds__(256)
vq_reduce_loss(const float* __restrict__ partials, float* __restrict__ out_loss) {
  __shared__ float red[8];
  float s = 0.f;
  for (int i = threadIdx.x; i < 4096; i += 256) s += partials[i];
#pragma unroll
  for (int m = 16; m >= 1; m >>= 1) s += __shfl_xor(s, m, 32);
  if ((threadIdx.x & 31) == 0) red[threadIdx.x >> 5] = s;
  __syncthreads();
  if (threadIdx.x == 0) {
    float tot = 0.f;
#pragma unroll
    for (int w = 0; w < 8; ++w) tot += red[w];
    *out_loss = tot / (float)BCT;
  }
}

// ---------------------------------------------------------------------------
extern "C" void kernel_launch(void* const* d_in, const int* in_sizes, int n_in,
                              void* d_out, int out_size, void* d_ws, size_t ws_size,
                              hipStream_t stream) {
  const float* z  = (const float*)d_in[0];   // [B, C, T] f32
  const float* cb = (const float*)d_in[1];   // [K, C]    f32
  float* out = (float*)d_out;                // [BCT zq | BT idx | 1 loss]
  char* ws = (char*)d_ws;

  // Workspace layout (~8.2 MB total)
  unsigned short* cb_hi = (unsigned short*)(ws);                       // 4 MB
  unsigned short* cb_lo = (unsigned short*)(ws + (size_t)4194304);     // 4 MB
  float* norms = (float*)(ws + (size_t)8388608);                       // 32 KB
  int*   idx   = (int*)  (ws + (size_t)8421376);                       // 128 KB
  float* parts = (float*)(ws + (size_t)8552448);                       // 16 KB

  vq_prep_codebook<<<KCODES / 8, 256, 0, stream>>>(cb, cb_hi, cb_lo, norms);
  vq_argmin<<<NROWS / 128, 256, 0, stream>>>(z, cb_hi, cb_lo, norms, idx);
  vq_gather<<<4096, 256, 0, stream>>>(z, cb, idx, out, out + BCT, parts);
  vq_reduce_loss<<<1, 256, 0, stream>>>(parts, out + BCT + NROWS);
}
